// Attention_87668872446125
// MI455X (gfx1250) — compile-verified, matrix operand order checked by reading, not run on hardware
//
#include <hip/hip_runtime.h>
#include <hip/hip_bf16.h>

typedef _Float16 h8  __attribute__((ext_vector_type(8)));
typedef _Float16 h16 __attribute__((ext_vector_type(16)));
typedef float    f8  __attribute__((ext_vector_type(8)));

#define DIM      1024
#define NSEQ     2048
#define BATCH    2
#define HEADS    16
#define HDIM     64
#define MROWS    (BATCH * NSEQ)   // 4096

__device__ __forceinline__ f8 wmma_f16(h16 a, h16 b, f8 c) {
    // v_wmma_f32_16x16x32_f16: D = A(16x32 f16) * B(32x16 f16) + C(16x16 f32)
    return __builtin_amdgcn_wmma_f32_16x16x32_f16(
        /*neg_a=*/false, a, /*neg_b=*/false, b,
        /*c_mod=*/(short)0, c, /*reuse_a=*/false, /*reuse_b=*/false);
}

// Loads a 16x32 f16 operand fragment (A rows or B columns; layouts symmetric):
// lane = {half(1b), idx(4b)}; lane reads row/col `idx` (leading dim ld),
// 8 halves at k0 + half*8 and 8 halves at k0 + 16 + half*8 (two 16B loads).
__device__ __forceinline__ h16 load_frag(const _Float16* __restrict__ base,
                                         int ld, int k0) {
    int lane = threadIdx.x & 31;
    int half = lane >> 4;
    int idx  = lane & 15;
    const _Float16* p = base + (size_t)idx * (size_t)ld + k0 + half * 8;
    h8 lo = *(const h8*)p;
    h8 hi = *(const h8*)(p + 16);
    return __builtin_shufflevector(lo, hi, 0,1,2,3,4,5,6,7,8,9,10,11,12,13,14,15);
}

// 64x64 per-wave GEMM tile core: 16 WMMAs per 8 fragment loads per k-step.
__device__ __forceinline__ void gemm_tile64(const _Float16* __restrict__ Abase,
                                            const _Float16* __restrict__ Bbase,
                                            int lda, int ldb, int K,
                                            f8 acc[4][4]) {
    for (int k0 = 0; k0 < K; k0 += 32) {
        h16 af[4], bf[4];
        #pragma unroll
        for (int i = 0; i < 4; ++i)
            af[i] = load_frag(Abase + (size_t)(i * 16) * lda, lda, k0);
        #pragma unroll
        for (int j = 0; j < 4; ++j)
            bf[j] = load_frag(Bbase + (size_t)(j * 16) * ldb, ldb, k0);
        #pragma unroll
        for (int i = 0; i < 4; ++i)
            #pragma unroll
            for (int j = 0; j < 4; ++j)
                acc[i][j] = wmma_f16(af[i], bf[j], acc[i][j]);
    }
}

// ---------------------------------------------------------------- conversions
__global__ void __launch_bounds__(256) f32_to_f16_kernel(
        const float* __restrict__ in, _Float16* __restrict__ out, int n8) {
    int i = blockIdx.x * blockDim.x + threadIdx.x;
    int stride = gridDim.x * blockDim.x;
    for (; i < n8; i += stride) {
        const float* p = in + (size_t)i * 8;
        h8 v;
        #pragma unroll
        for (int j = 0; j < 8; ++j) v[j] = (_Float16)p[j];
        *(h8*)(out + (size_t)i * 8) = v;
    }
}

// ---------------------------------------------------------------- QKV GEMM
// y[m,o] = sum_k x16[m,k] * wq16[o,k]  (m in [0,4096), o in [0,3072)).
// Wave owns a 64x64 tile. A 64-wide column tile never straddles a 1024
// boundary, so which-of-{q,k,v}, head index and batch index are WAVE-UNIFORM:
// compute them from tile coords so the epilogue uses scalar branches only.
__global__ void __launch_bounds__(256) gemm_qkv_kernel(
        const _Float16* __restrict__ x16, const _Float16* __restrict__ wq16,
        _Float16* __restrict__ q16, _Float16* __restrict__ k16,
        _Float16* __restrict__ vT16) {
    const int NT = (3 * DIM) / 64;                       // 48 col tiles
    int wid = blockIdx.x * (blockDim.x >> 5) + (threadIdx.x >> 5);
    int mt = wid / NT, nt = wid % NT;
    int m0 = mt * 64, n0 = nt * 64;

    f8 acc[4][4] = {};
    gemm_tile64(x16 + (size_t)m0 * DIM, wq16 + (size_t)n0 * DIM, DIM, DIM, DIM, acc);

    int lane = threadIdx.x & 31, half = lane >> 4, idx = lane & 15;
    // uniform tile decomposition
    int which = n0 >> 10;                    // 0=q 1=k 2=v (uniform)
    int h     = (n0 & (DIM - 1)) >> 6;       // head (uniform: (n0&63)==0)
    int b     = m0 >> 11;                    // batch (uniform: 64-row tiles)
    int nrow0 = m0 & (NSEQ - 1);             // seq base (uniform)
    size_t bh = (size_t)(b * HEADS + h);

    #pragma unroll
    for (int ni = 0; ni < 4; ++ni) {
        int d = ((n0 & 63) + ni * 16) + idx; // 0..63 within head
        #pragma unroll
        for (int mi = 0; mi < 4; ++mi) {
            int nbase = nrow0 + mi * 16 + half * 8;      // 8 consecutive n's
            if (which == 2) {
                // V^T [B,H,D,N]: contiguous in n -> one 16B store per frag
                h8 v;
                #pragma unroll
                for (int i = 0; i < 8; ++i) v[i] = (_Float16)acc[mi][ni][i];
                *(h8*)(vT16 + (bh * HDIM + d) * NSEQ + nbase) = v;
            } else {
                _Float16* dst = (which == 0 ? q16 : k16)
                              + ((bh * NSEQ + nbase) * HDIM + d);
                #pragma unroll
                for (int i = 0; i < 8; ++i)
                    dst[(size_t)i * HDIM] = (_Float16)acc[mi][ni][i];
            }
        }
    }
}

// ---------------------------------------------------------------- attention
// One wave per (b,h, 32 query rows): two 16-query groups share every K / V^T
// fragment -> 16 WMMAs per 8 fragment loads per 32-kv tile.
// S^T = K * Q^T: each lane owns one query column, so softmax stats are
// per-lane + one shfl_xor(16). The f32 C-layout of the stacked S^T frags IS
// the f16 B-operand layout for O^T = V^T * P^T (lane-local P conversion).
__global__ void __launch_bounds__(256) attn_kernel(
        const _Float16* __restrict__ q16, const _Float16* __restrict__ k16,
        const _Float16* __restrict__ vT16, _Float16* __restrict__ o16) {
    int wid = blockIdx.x * (blockDim.x >> 5) + (threadIdx.x >> 5);
    int qt = wid & 63;                  // 64 query tiles of 32 rows
    int bh = wid >> 6;                  // 0..31
    const _Float16* Q  = q16  + (size_t)bh * NSEQ * HDIM;
    const _Float16* K  = k16  + (size_t)bh * NSEQ * HDIM;
    const _Float16* VT = vT16 + (size_t)bh * HDIM * NSEQ;

    // Q^T B-fragments for both query groups, pre-scaled by 1/sqrt(D)
    const _Float16 scale = (_Float16)0.125f;     // 1/sqrt(64)
    h16 qf[2][2];
    #pragma unroll
    for (int g = 0; g < 2; ++g) {
        const _Float16* Qg = Q + (size_t)(qt * 32 + g * 16) * HDIM;
        qf[g][0] = load_frag(Qg, HDIM, 0)  * scale;
        qf[g][1] = load_frag(Qg, HDIM, 32) * scale;
    }

    f8 oacc[2][4] = {};                          // O^T accum per group
    float mrow[2] = {-1e30f, -1e30f};
    float lrow[2] = {0.0f, 0.0f};

    for (int kv0 = 0; kv0 < NSEQ; kv0 += 32) {
        // shared K fragments: [kv-half][k-step]
        h16 ak[2][2];
        ak[0][0] = load_frag(K + (size_t)kv0        * HDIM, HDIM, 0);
        ak[0][1] = load_frag(K + (size_t)kv0        * HDIM, HDIM, 32);
        ak[1][0] = load_frag(K + (size_t)(kv0 + 16) * HDIM, HDIM, 0);
        ak[1][1] = load_frag(K + (size_t)(kv0 + 16) * HDIM, HDIM, 32);
        // shared V^T fragments: 16d x 32kv each
        h16 av[4];
        #pragma unroll
        for (int df = 0; df < 4; ++df)
            av[df] = load_frag(VT + (size_t)(df * 16) * NSEQ, NSEQ, kv0);

        #pragma unroll
        for (int g = 0; g < 2; ++g) {
            f8 s0 = {}; s0 = wmma_f16(ak[0][0], qf[g][0], s0);
                        s0 = wmma_f16(ak[0][1], qf[g][1], s0);
            f8 s1 = {}; s1 = wmma_f16(ak[1][0], qf[g][0], s1);
                        s1 = wmma_f16(ak[1][1], qf[g][1], s1);

            // online softmax: per-lane over 16 kv values, xor-16 merges halves
            float mt = s0[0];
            #pragma unroll
            for (int j = 1; j < 8; ++j) mt = fmaxf(mt, s0[j]);
            #pragma unroll
            for (int j = 0; j < 8; ++j) mt = fmaxf(mt, s1[j]);
            mt = fmaxf(mt, __shfl_xor(mt, 16, 32));
            float mnew  = fmaxf(mrow[g], mt);
            float alpha = __expf(mrow[g] - mnew);
            mrow[g] = mnew;

            h16 pv;                               // P^T B-operand, f16
            float lt = 0.0f;
            #pragma unroll
            for (int j = 0; j < 8; ++j) {
                float p = __expf(s0[j] - mnew); lt += p; pv[j]     = (_Float16)p;
            }
            #pragma unroll
            for (int j = 0; j < 8; ++j) {
                float p = __expf(s1[j] - mnew); lt += p; pv[8 + j] = (_Float16)p;
            }
            lt += __shfl_xor(lt, 16, 32);
            lrow[g] = lrow[g] * alpha + lt;

            #pragma unroll
            for (int df = 0; df < 4; ++df) {
                oacc[g][df] = oacc[g][df] * alpha;
                oacc[g][df] = wmma_f16(av[df], pv, oacc[g][df]);
            }
        }
    }

    // store O (f16) in [B, N, C] layout; 16B per lane per frag
    int lane = threadIdx.x & 31, half = lane >> 4, idx = lane & 15;
    int b = bh >> 4, h = bh & 15;
    #pragma unroll
    for (int g = 0; g < 2; ++g) {
        int n = qt * 32 + g * 16 + idx;
        float rl = 1.0f / lrow[g];
        #pragma unroll
        for (int df = 0; df < 4; ++df) {
            h8 outv;
            #pragma unroll
            for (int i = 0; i < 8; ++i) outv[i] = (_Float16)(oacc[g][df][i] * rl);
            size_t off = ((size_t)(b * NSEQ + n)) * DIM + h * HDIM + df * 16 + half * 8;
            *(h8*)(o16 + off) = outv;
        }
    }
}

// ---------------------------------------------------------------- proj GEMM
// out[m,o] = sum_k o16[m,k] * wp16[o,k] + bias[o], f32 output.
// Stores: per instruction, each 16-lane half writes 64B contiguous (one row
// segment), so the scattered-looking epilogue coalesces per cacheline.
__global__ void __launch_bounds__(256) gemm_proj_kernel(
        const _Float16* __restrict__ o16, const _Float16* __restrict__ wp16,
        const float* __restrict__ bias, float* __restrict__ out) {
    const int NT = DIM / 64;                         // 16 col tiles
    int wid = blockIdx.x * (blockDim.x >> 5) + (threadIdx.x >> 5);
    int mt = wid / NT, nt = wid % NT;
    int m0 = mt * 64, n0 = nt * 64;

    f8 acc[4][4] = {};
    gemm_tile64(o16 + (size_t)m0 * DIM, wp16 + (size_t)n0 * DIM, DIM, DIM, DIM, acc);

    int lane = threadIdx.x & 31, half = lane >> 4, idx = lane & 15;
    #pragma unroll
    for (int ni = 0; ni < 4; ++ni) {
        int o = n0 + ni * 16 + idx;
        float bv = bias[o];
        #pragma unroll
        for (int mi = 0; mi < 4; ++mi) {
            int mbase = m0 + mi * 16 + half * 8;
            float* dst = out + (size_t)mbase * DIM + o;
            #pragma unroll
            for (int i = 0; i < 8; ++i)
                dst[(size_t)i * DIM] = acc[mi][ni][i] + bv;
        }
    }
}

// ---------------------------------------------------------------- launcher
extern "C" void kernel_launch(void* const* d_in, const int* in_sizes, int n_in,
                              void* d_out, int out_size, void* d_ws, size_t ws_size,
                              hipStream_t stream) {
    const float* x      = (const float*)d_in[0];   // [2,2048,1024]
    const float* w_qkv  = (const float*)d_in[1];   // [3072,1024]
    const float* w_proj = (const float*)d_in[2];   // [1024,1024]
    const float* b_proj = (const float*)d_in[3];   // [1024]
    float* out = (float*)d_out;                    // [2,2048,1024] f32

    size_t off = 0;
    char* base = (char*)d_ws;
    auto alloc = [&](size_t bytes) -> void* {
        void* p = base + off;
        off += (bytes + 255) & ~(size_t)255;
        return p;
    };
    _Float16* x16  = (_Float16*)alloc((size_t)MROWS * DIM * 2);         // 8 MB
    _Float16* wq16 = (_Float16*)alloc((size_t)3 * DIM * DIM * 2);       // 6 MB
    _Float16* wp16 = (_Float16*)alloc((size_t)DIM * DIM * 2);           // 2 MB
    _Float16* q16  = (_Float16*)alloc((size_t)MROWS * DIM * 2);         // 8 MB
    _Float16* k16  = (_Float16*)alloc((size_t)MROWS * DIM * 2);         // 8 MB
    _Float16* vT16 = (_Float16*)alloc((size_t)MROWS * DIM * 2);         // 8 MB
    _Float16* o16  = (_Float16*)alloc((size_t)MROWS * DIM * 2);         // 8 MB

    f32_to_f16_kernel<<<1024, 256, 0, stream>>>(x,      x16,  (MROWS * DIM) / 8);
    f32_to_f16_kernel<<<1024, 256, 0, stream>>>(w_qkv,  wq16, (3 * DIM * DIM) / 8);
    f32_to_f16_kernel<<<512,  256, 0, stream>>>(w_proj, wp16, (DIM * DIM) / 8);

    // QKV: 64 x 48 tiles of 64x64 -> 3072 waves -> 384 blocks
    gemm_qkv_kernel<<<384, 256, 0, stream>>>(x16, wq16, q16, k16, vT16);

    // attention: 32 (b,h) x 64 q-tiles = 2048 waves -> 256 blocks
    attn_kernel<<<256, 256, 0, stream>>>(q16, k16, vT16, o16);

    // proj: 64 x 16 tiles -> 1024 waves -> 128 blocks
    gemm_proj_kernel<<<128, 256, 0, stream>>>(o16, wp16, b_proj, out);
}
